// KalmanNetNN_13314398618405
// MI455X (gfx1250) — matrix operand order repaired and checked by reading, not compiled, not run on hardware
//
#include <hip/hip_runtime.h>
#include <hip/hip_bf16.h>

typedef _Float16 half_t;
typedef __attribute__((ext_vector_type(16))) _Float16 v16h;
typedef __attribute__((ext_vector_type(8)))  float    v8f;

#define WAVES 2
#define ROWS_PER_WAVE 32
#define ROWS_PER_BLOCK (WAVES*ROWS_PER_WAVE)
#define B_TOTAL 65536
#define WCH 13312   // per-wave LDS halfs

// workspace offsets (in halfs); each weight stored as pre-swizzled f16 B-fragments:
// frag(nt,kb) = 32 lanes x 16 halfs contiguous; elem e of lane L:
//   N = nt*16 + (L&15), K = kb*32 + (e&7) + 8*(L>>4) + 16*(e>>3); zero-padded OOB.
enum : int {
  OFF_FC5    = 0,
  OFF_FC6    = OFF_FC5    + 1*3*512,
  OFF_FC7    = OFF_FC6    + 1*3*512,
  OFF_GQIH   = OFF_FC7    + 1*5*512,
  OFF_GQHH   = OFF_GQIH   + 2*12*512,
  OFF_GSIGIH = OFF_GQHH   + 2*12*512,
  OFF_GSIGHH = OFF_GSIGIH + 4*12*512,
  OFF_GSIH   = OFF_GSIGHH + 2*12*512,
  OFF_GSHH   = OFF_GSIH   + 5*12*512,
  OFF_FC1    = OFF_GSHH   + 2*12*512,
  OFF_FC2W1  = OFF_FC1    + 2*4*512,
  OFF_FC2W2  = OFF_FC2W1  + 4*32*512,
  OFF_END    = OFF_FC2W2  + 16*4*512
};

__device__ __forceinline__ v8f wmma16(v16h a, v16h b, v8f c){
  return __builtin_amdgcn_wmma_f32_16x16x32_f16(false, a, false, b, (short)0, c, false, false);
}
__device__ __forceinline__ v8f zero8(){ v8f z = {0.f,0.f,0.f,0.f,0.f,0.f,0.f,0.f}; return z; }

// B-fragment: two b128 loads per lane from pre-swizzled workspace
__device__ __forceinline__ v16h ldB(const half_t* w, int KB, int nt, int kb, int lane){
  return *reinterpret_cast<const v16h*>(w + (((nt*KB + kb) << 9) + (lane << 4)));
}
// A-fragment from LDS activation buffer [16 rows][W cols] (buf pre-offset to tile), f16
// A layout: lane L row = L&15; elem e -> col k0 + (e&7) + 16*(e>>3) + 8*(L>>4)
__device__ __forceinline__ v16h ldsA(const half_t* buf, int W, int lane, int k0){
  const half_t* p = buf + (lane & 15)*W + k0 + ((lane >> 4) << 3);
  v16h a;
  #pragma unroll
  for(int i=0;i<8;i++){ a[i] = p[i]; a[i+8] = p[16+i]; }
  return a;
}
// A-fragment from global f32 hidden state [B][64]
__device__ __forceinline__ v16h ldAg(const float* h, int rowBase, int lane, int k0){
  const float* p = h + (size_t)(rowBase + (lane & 15))*64 + k0 + ((lane >> 4) << 3);
  v16h a;
  #pragma unroll
  for(int i=0;i<8;i++){ a[i] = (half_t)p[i]; a[i+8] = (half_t)p[16+i]; }
  return a;
}
// D-layout (C/D: lane holds N=n0+(L&15), rows r+8*(L>>4)) elementwise load of hidden
__device__ __forceinline__ v8f ldDg(const float* h, int rowBase, int lane, int n0){
  int col = n0 + (lane & 15);
  int rb  = (lane >> 4) << 3;
  v8f v;
  #pragma unroll
  for(int r=0;r<8;r++) v[r] = h[(size_t)(rowBase + rb + r)*64 + col];
  return v;
}
// store D tile into LDS activation buffer as f16 (buf pre-offset to tile)
__device__ __forceinline__ void stD(half_t* buf, int W, int lane, int n0, v8f d){
  int col = n0 + (lane & 15);
  int rb  = (lane >> 4) << 3;
  #pragma unroll
  for(int r=0;r<8;r++) buf[(rb + r)*W + col] = (half_t)d[r];
}
__device__ __forceinline__ void zfill(half_t* buf, int W, int lane, int c0, int c1){
  int span = c1 - c0;
  for(int i = lane; i < 32*span; i += 32){
    int r = i / span, c = c0 + (i - r*span);
    buf[r*W + c] = (half_t)0.f;
  }
}
__device__ __forceinline__ float sigm(float x){ return 1.f/(1.f + __expf(-x)); }
__device__ __forceinline__ void norm8(float* v){
  float s = 0.f;
  #pragma unroll
  for(int i=0;i<8;i++) s += v[i]*v[i];
  float n = fmaxf(sqrtf(s), 1e-12f);
  float inv = 1.f / n;
  #pragma unroll
  for(int i=0;i<8;i++) v[i] *= inv;
}

// FC + relu on two 16-row tiles; B fragments loaded once, used twice.
template<int KB, int NT>
__device__ __forceinline__ void fc_relu2(const v16h (&A)[2][KB], const half_t* w,
                                         const float* bias, int dout,
                                         half_t* dst, int dW, int c0, int lane){
  int nl = lane & 15;
  #pragma unroll
  for(int nt=0; nt<NT; nt++){
    v8f a0 = zero8(), a1 = zero8();
    #pragma unroll
    for(int kb=0; kb<KB; kb++){
      v16h b = ldB(w, KB, nt, kb, lane);
      a0 = wmma16(A[0][kb], b, a0);
      a1 = wmma16(A[1][kb], b, a1);
    }
    int n = nt*16 + nl;
    float bb = (n < dout) ? bias[n] : 0.f;
    v8f d0, d1;
    #pragma unroll
    for(int j=0;j<8;j++){
      float v0 = a0[j] + bb; d0[j] = v0 > 0.f ? v0 : 0.f;
      float v1 = a1[j] + bb; d1[j] = v1 > 0.f ? v1 : 0.f;
    }
    stD(dst,           dW, lane, c0 + nt*16, d0);
    stD(dst + 16*dW,   dW, lane, c0 + nt*16, d1);
  }
}

// GRU cell (gate order r,z,n) on two 16-row tiles; per-column-tile gate fusion
// keeps r/z transient. For r,z the x-part and h-part share one accumulator.
template<int KBX>
__device__ __forceinline__ void gru2(const v16h (&AX)[2][KBX], const v16h (&AH)[2][2],
                                     const half_t* wih, const half_t* whh,
                                     const float* bih, const float* bhh,
                                     const float* hptr, int rowBase, int lane,
                                     half_t* dst, int dW, int c0){
  int nl = lane & 15;
  #pragma unroll
  for(int t=0;t<4;t++){
    // ---- r gate (tile t) ----
    v8f r0, r1;
    {
      v8f g0 = zero8(), g1 = zero8();
      #pragma unroll
      for(int kb=0;kb<KBX;kb++){
        v16h b = ldB(wih, KBX, t, kb, lane);
        g0 = wmma16(AX[0][kb], b, g0); g1 = wmma16(AX[1][kb], b, g1);
      }
      #pragma unroll
      for(int kb=0;kb<2;kb++){
        v16h b = ldB(whh, 2, t, kb, lane);
        g0 = wmma16(AH[0][kb], b, g0); g1 = wmma16(AH[1][kb], b, g1);
      }
      float bs = bih[t*16 + nl] + bhh[t*16 + nl];
      #pragma unroll
      for(int j=0;j<8;j++){ r0[j] = sigm(g0[j] + bs); r1[j] = sigm(g1[j] + bs); }
    }
    // ---- z gate (tile 4+t) ----
    v8f z0, z1;
    {
      v8f g0 = zero8(), g1 = zero8();
      #pragma unroll
      for(int kb=0;kb<KBX;kb++){
        v16h b = ldB(wih, KBX, 4+t, kb, lane);
        g0 = wmma16(AX[0][kb], b, g0); g1 = wmma16(AX[1][kb], b, g1);
      }
      #pragma unroll
      for(int kb=0;kb<2;kb++){
        v16h b = ldB(whh, 2, 4+t, kb, lane);
        g0 = wmma16(AH[0][kb], b, g0); g1 = wmma16(AH[1][kb], b, g1);
      }
      float bs = bih[64 + t*16 + nl] + bhh[64 + t*16 + nl];
      #pragma unroll
      for(int j=0;j<8;j++){ z0[j] = sigm(g0[j] + bs); z1[j] = sigm(g1[j] + bs); }
    }
    // ---- n gate + output (tile 8+t); x-part and h-part kept separate ----
    {
      v8f gi0 = zero8(), gi1 = zero8(), gh0 = zero8(), gh1 = zero8();
      #pragma unroll
      for(int kb=0;kb<KBX;kb++){
        v16h b = ldB(wih, KBX, 8+t, kb, lane);
        gi0 = wmma16(AX[0][kb], b, gi0); gi1 = wmma16(AX[1][kb], b, gi1);
      }
      #pragma unroll
      for(int kb=0;kb<2;kb++){
        v16h b = ldB(whh, 2, 8+t, kb, lane);
        gh0 = wmma16(AH[0][kb], b, gh0); gh1 = wmma16(AH[1][kb], b, gh1);
      }
      v8f hv0 = ldDg(hptr, rowBase,      lane, t*16);
      v8f hv1 = ldDg(hptr, rowBase + 16, lane, t*16);
      float bi = bih[128 + t*16 + nl], bh = bhh[128 + t*16 + nl];
      v8f o0, o1;
      #pragma unroll
      for(int j=0;j<8;j++){
        float n0 = tanhf(gi0[j] + bi + r0[j]*(gh0[j] + bh));
        float n1 = tanhf(gi1[j] + bi + r1[j]*(gh1[j] + bh));
        o0[j] = (1.f - z0[j])*n0 + z0[j]*hv0[j];
        o1[j] = (1.f - z1[j])*n1 + z1[j]*hv1[j];
      }
      stD(dst,         dW, lane, c0 + t*16, o0);
      stD(dst + 16*dW, dW, lane, c0 + t*16, o1);
    }
  }
}

struct KParams {
  const float *y, *post, *postPrev, *priorPrev, *yPrev;
  const float *hQ, *hSig, *hS, *F, *H;
  const float *bQi, *bQh, *bSigi, *bSigh, *bSi, *bSh;
  const float *fc1b, *fc2b1, *fc2b2, *fc5b, *fc6b, *fc7b;
  const half_t* w;
  float* out;
};

__global__ void __launch_bounds__(WAVES*32)
knet_main(KParams p){
  extern __shared__ half_t smem[];
  const int lane = threadIdx.x & 31;
  const int wave = threadIdx.x >> 5;
  half_t* base  = smem + wave*WCH;
  half_t* featE = base;            // 32x32
  half_t* featU = base + 1024;     // 32x32
  half_t* featO = base + 2048;     // 32x32
  half_t* xbuf  = base + 3072;     // 32x160
  half_t* ybuf  = base + 8192;     // 32x128 (persist out_Sigma | out_S)
  half_t* stg   = base + 12288;    // 32x32
  const int rowBase = blockIdx.x * ROWS_PER_BLOCK + wave * ROWS_PER_WAVE;
  const int nl  = lane & 15;
  const half_t* w = p.w;

  // ---- per-row features: one row per lane (32 rows/wave) ----
  float yv[8], post_[8], prior_[8], m1y[8], dy[8];
  {
    const float* yp = p.y    + (size_t)(rowBase+lane)*8;
    const float* pp = p.post + (size_t)(rowBase+lane)*8;
    #pragma unroll
    for(int i=0;i<8;i++){ yv[i]=yp[i]; post_[i]=pp[i]; }
  }
  #pragma unroll
  for(int i=0;i<8;i++){ float s=0.f;
    #pragma unroll
    for(int j=0;j<8;j++) s += p.F[i*8+j]*post_[j];
    prior_[i]=s; }
  #pragma unroll
  for(int i=0;i<8;i++){ float s=0.f;
    #pragma unroll
    for(int j=0;j<8;j++) s += p.H[i*8+j]*prior_[j];
    m1y[i]=s; }
  {
    const float* ypv = p.yPrev     + (size_t)(rowBase+lane)*8;
    const float* ppv = p.postPrev  + (size_t)(rowBase+lane)*8;
    const float* prv = p.priorPrev + (size_t)(rowBase+lane)*8;
    float od[8], oid[8], fe[8], fu[8];
    #pragma unroll
    for(int i=0;i<8;i++){
      od[i]  = yv[i] - ypv[i];
      dy[i]  = yv[i] - m1y[i];
      oid[i] = dy[i];
      fe[i]  = post_[i] - ppv[i];
      fu[i]  = post_[i] - prv[i];
    }
    norm8(od); norm8(oid); norm8(fe); norm8(fu);
    #pragma unroll
    for(int i=0;i<8;i++){
      featO[lane*32 + i]     = (half_t)od[i];
      featO[lane*32 + 8 + i] = (half_t)oid[i];
      featE[lane*32 + i]     = (half_t)fe[i];
      featU[lane*32 + i]     = (half_t)fu[i];
    }
    zfill(featE, 32, lane, 8, 32);
    zfill(featU, 32, lane, 8, 32);
    zfill(featO, 32, lane, 16, 32);
  }

  // ---- FC5: feat(8,pad32) -> 40, relu; into xbuf[0:64) (pad zeros) ----
  { v16h A[2][1] = { { ldsA(featE, 32, lane, 0) }, { ldsA(featE + 512, 32, lane, 0) } };
    fc_relu2<1,3>(A, w + OFF_FC5, p.fc5b, 40, xbuf, 160, 0, lane); }
  zfill(xbuf, 160, lane, 48, 64);

  // ---- GRU Q: x=xbuf[0:64), h=hQ -> out_Q into xbuf[0:64) ----
  { v16h AX[2][2], AH[2][2];
    #pragma unroll
    for(int tt=0;tt<2;tt++){
      #pragma unroll
      for(int kb=0;kb<2;kb++){
        AX[tt][kb] = ldsA(xbuf + tt*16*160, 160, lane, kb*32);
        AH[tt][kb] = ldAg(p.hQ, rowBase + tt*16, lane, kb*32);
      }
    }
    gru2<2>(AX, AH, w+OFF_GQIH, w+OFF_GQHH, p.bQi, p.bQh, p.hQ, rowBase, lane, xbuf, 160, 0); }

  // ---- FC6 -> xbuf[64:112), pad to 128 ----
  { v16h A[2][1] = { { ldsA(featU, 32, lane, 0) }, { ldsA(featU + 512, 32, lane, 0) } };
    fc_relu2<1,3>(A, w + OFF_FC6, p.fc6b, 40, xbuf, 160, 64, lane); }
  zfill(xbuf, 160, lane, 112, 128);

  // ---- GRU Sigma: x=xbuf[0:128), h=hSig -> out_Sigma into ybuf[0:64) ----
  { v16h AX[2][4], AH[2][2];
    #pragma unroll
    for(int tt=0;tt<2;tt++){
      #pragma unroll
      for(int kb=0;kb<4;kb++) AX[tt][kb] = ldsA(xbuf + tt*16*160, 160, lane, kb*32);
      #pragma unroll
      for(int kb=0;kb<2;kb++) AH[tt][kb] = ldAg(p.hSig, rowBase + tt*16, lane, kb*32);
    }
    gru2<4>(AX, AH, w+OFF_GSIGIH, w+OFF_GSIGHH, p.bSigi, p.bSigh, p.hSig, rowBase, lane, ybuf, 128, 0); }

  // ---- FC1: out_Sigma -> 64, relu; into xbuf[0:64) ----
  { v16h A[2][2];
    #pragma unroll
    for(int tt=0;tt<2;tt++){
      A[tt][0] = ldsA(ybuf + tt*16*128, 128, lane, 0);
      A[tt][1] = ldsA(ybuf + tt*16*128, 128, lane, 32);
    }
    fc_relu2<2,4>(A, w + OFF_FC1, p.fc1b, 64, xbuf, 160, 0, lane); }

  // ---- FC7: [obs_diff,obs_innov] (16,pad32) -> 80, relu; xbuf[64:144), pad 160 ----
  { v16h A[2][1] = { { ldsA(featO, 32, lane, 0) }, { ldsA(featO + 512, 32, lane, 0) } };
    fc_relu2<1,5>(A, w + OFF_FC7, p.fc7b, 80, xbuf, 160, 64, lane); }
  zfill(xbuf, 160, lane, 144, 160);

  // ---- GRU S: x=xbuf[0:160), h=hS -> out_S into ybuf[64:128) ----
  { v16h AX[2][5], AH[2][2];
    #pragma unroll
    for(int tt=0;tt<2;tt++){
      #pragma unroll
      for(int kb=0;kb<5;kb++) AX[tt][kb] = ldsA(xbuf + tt*16*160, 160, lane, kb*32);
      #pragma unroll
      for(int kb=0;kb<2;kb++) AH[tt][kb] = ldAg(p.hS, rowBase + tt*16, lane, kb*32);
    }
    gru2<5>(AX, AH, w+OFF_GSIH, w+OFF_GSHH, p.bSi, p.bSh, p.hS, rowBase, lane, ybuf, 128, 64); }

  // ---- FC2 fused: hidden(512) in 32-col chunks via stg, accumulate out(64) ----
  v16h aF[2][4];
  #pragma unroll
  for(int tt=0;tt<2;tt++)
    #pragma unroll
    for(int kb=0;kb<4;kb++) aF[tt][kb] = ldsA(ybuf + tt*16*128, 128, lane, kb*32);
  v8f acc2[2][4];
  #pragma unroll
  for(int t=0;t<4;t++){ acc2[0][t] = zero8(); acc2[1][t] = zero8(); }
  for(int hb=0; hb<16; hb++){
    #pragma unroll
    for(int s=0;s<2;s++){
      int nt = hb*2 + s;
      v8f d0 = zero8(), d1 = zero8();
      #pragma unroll
      for(int kb=0;kb<4;kb++){
        v16h b = ldB(w + OFF_FC2W1, 4, nt, kb, lane);
        d0 = wmma16(aF[0][kb], b, d0);
        d1 = wmma16(aF[1][kb], b, d1);
      }
      float bb = p.fc2b1[nt*16 + nl];
      v8f r0, r1;
      #pragma unroll
      for(int j=0;j<8;j++){
        float v0 = d0[j] + bb; r0[j] = v0 > 0.f ? v0 : 0.f;
        float v1 = d1[j] + bb; r1[j] = v1 > 0.f ? v1 : 0.f;
      }
      stD(stg,       32, lane, s*16, r0);
      stD(stg + 512, 32, lane, s*16, r1);
    }
    v16h ah0 = ldsA(stg,       32, lane, 0);
    v16h ah1 = ldsA(stg + 512, 32, lane, 0);
    #pragma unroll
    for(int t=0;t<4;t++){
      v16h b = ldB(w + OFF_FC2W2, 16, t, hb, lane);
      acc2[0][t] = wmma16(ah0, b, acc2[0][t]);
      acc2[1][t] = wmma16(ah1, b, acc2[1][t]);
    }
  }
  #pragma unroll
  for(int t=0;t<4;t++){
    float bb = p.fc2b2[t*16 + nl];
    v8f d0, d1;
    #pragma unroll
    for(int j=0;j<8;j++){ d0[j] = acc2[0][t][j] + bb; d1[j] = acc2[1][t][j] + bb; }
    stD(xbuf,        64, lane, t*16, d0);   // KGain flat, rows 0..15
    stD(xbuf + 1024, 64, lane, t*16, d1);   // rows 16..31
  }

  // ---- posterior_new = prior + KGain @ dy (one row per lane) ----
  {
    #pragma unroll
    for(int i=0;i<8;i++){
      float s = prior_[i];
      #pragma unroll
      for(int j=0;j<8;j++) s += (float)xbuf[lane*64 + i*8 + j] * dy[j];
      p.out[(size_t)(rowBase+lane)*8 + i] = s;
    }
  }
}

// convert fp32 weight [dout][din] into f16 B-fragment layout (zero padded)
__global__ void prep_weight(const float* __restrict__ src, half_t* __restrict__ dst,
                            int din, int dout, int KB, int NT){
  int total = NT*KB*512;
  for(int i = blockIdx.x*blockDim.x + threadIdx.x; i < total; i += gridDim.x*blockDim.x){
    int frag = i >> 9, off = i & 511;
    int lane = off >> 4, e = off & 15;
    int nt = frag / KB, kb = frag - nt*KB;
    int n = nt*16 + (lane & 15);
    int k = kb*32 + (e & 7) + ((lane >> 4) << 3) + ((e >> 3) << 4);
    dst[i] = (half_t)((n < dout && k < din) ? src[n*din + k] : 0.f);
  }
}

extern "C" void kernel_launch(void* const* d_in, const int* in_sizes, int n_in,
                              void* d_out, int out_size, void* d_ws, size_t ws_size,
                              hipStream_t stream){
  half_t* w = (half_t*)d_ws;
  auto prep = [&](int idx, int off, int din, int dout, int KB, int NT){
    int total = NT*KB*512;
    int blocks = (total + 255)/256; if(blocks > 256) blocks = 256;
    prep_weight<<<blocks, 256, 0, stream>>>((const float*)d_in[idx], w + off, din, dout, KB, NT);
  };
  prep(32, OFF_FC5,     8,  40,  1,  3);   // fc5_w
  prep(34, OFF_FC6,     8,  40,  1,  3);   // fc6_w
  prep(36, OFF_FC7,    16,  80,  1,  5);   // fc7_w
  prep(10, OFF_GQIH,   40, 192,  2, 12);   // gruQ_Wih
  prep(11, OFF_GQHH,   64, 192,  2, 12);   // gruQ_Whh
  prep(14, OFF_GSIGIH,104, 192,  4, 12);   // gruSig_Wih
  prep(15, OFF_GSIGHH, 64, 192,  2, 12);   // gruSig_Whh
  prep(18, OFF_GSIH,  144, 192,  5, 12);   // gruS_Wih
  prep(19, OFF_GSHH,   64, 192,  2, 12);   // gruS_Whh
  prep(22, OFF_FC1,    64,  64,  2,  4);   // fc1_w
  prep(24, OFF_FC2W1, 128, 512,  4, 32);   // fc2_w1
  prep(26, OFF_FC2W2, 512,  64, 16,  4);   // fc2_w2

  KParams kp;
  kp.y         = (const float*)d_in[0];
  kp.post      = (const float*)d_in[1];
  kp.postPrev  = (const float*)d_in[2];
  kp.priorPrev = (const float*)d_in[3];
  kp.yPrev     = (const float*)d_in[4];
  kp.hQ        = (const float*)d_in[5];
  kp.hSig      = (const float*)d_in[6];
  kp.hS        = (const float*)d_in[7];
  kp.F         = (const float*)d_in[8];
  kp.H         = (const float*)d_in[9];
  kp.bQi       = (const float*)d_in[12];
  kp.bQh       = (const float*)d_in[13];
  kp.bSigi     = (const float*)d_in[16];
  kp.bSigh     = (const float*)d_in[17];
  kp.bSi       = (const float*)d_in[20];
  kp.bSh       = (const float*)d_in[21];
  kp.fc1b      = (const float*)d_in[23];
  kp.fc2b1     = (const float*)d_in[25];
  kp.fc2b2     = (const float*)d_in[27];
  kp.fc5b      = (const float*)d_in[33];
  kp.fc6b      = (const float*)d_in[35];
  kp.fc7b      = (const float*)d_in[37];
  kp.w         = (const half_t*)d_ws;
  kp.out       = (float*)d_out;

  dim3 grid(B_TOTAL / ROWS_PER_BLOCK);
  dim3 block(WAVES * 32);
  size_t shmem = (size_t)WAVES * WCH * sizeof(half_t);  // 53248 B
  knet_main<<<grid, block, shmem, stream>>>(kp);
}
